// CQAttention_58067957842536
// MI455X (gfx1250) — matrix operand order repaired and checked by reading, not compiled, not run on hardware
//
#include <hip/hip_runtime.h>
#include <hip/hip_bf16.h>
#include <stdint.h>

#define B_   16
#define D_   512
#define LC_  2048
#define LQ_  512
#define NEG_INF (-1e30f)

typedef __attribute__((ext_vector_type(16))) _Float16 v16h;
typedef __attribute__((ext_vector_type(8)))  _Float16 v8h;
typedef __attribute__((ext_vector_type(8)))  float    v8f;

// ---------------- WMMA operand loaders (gfx1250 16x16x32 f16 layouts) --------
// A operand (M=16, K=32): lane L (<16): K 0..7 & 16..23 ; lane>=16: K 8..15 & 24..31
__device__ __forceinline__ v16h load_A16(const _Float16* p, int ld) {
  const int lane = threadIdx.x & 31;
  const int m    = lane & 15;
  const int kb   = (lane >> 4) << 3;
  const _Float16* row = p + (size_t)m * ld + kb;
  v16h a;
  ((v8h*)&a)[0] = *(const v8h*)(row);
  ((v8h*)&a)[1] = *(const v8h*)(row + 16);
  return a;
}
// B operand (K=32, N=16): lane L (<16): K 0..7 & 16..23 ; lane>=16: K 8..15 & 24..31, N = L%16.
// Memory layout must be [n][k] (N-major, K contiguous).
__device__ __forceinline__ v16h load_B16(const _Float16* p, int ld) {
  const int lane = threadIdx.x & 31;
  const int n    = lane & 15;
  const int kb   = (lane >> 4) << 3;
  const _Float16* col = p + (size_t)n * ld + kb;
  v16h b;
  ((v8h*)&b)[0] = *(const v8h*)(col);
  ((v8h*)&b)[1] = *(const v8h*)(col + 16);
  return b;
}
__device__ __forceinline__ v8f wmma16(v16h a, v16h b, v8f c) {
  return __builtin_amdgcn_wmma_f32_16x16x32_f16(false, a, false, b, (short)0, c, false, false);
}

// ---------------- Prep kernels ----------------------------------------------
// f32 -> f16 convert, same layout, 4 elems/thread
__global__ void k_conv_half(const float* __restrict__ in, _Float16* __restrict__ out) {
  size_t idx = ((size_t)blockIdx.x * blockDim.x + threadIdx.x) * 4;
  float4 v = *(const float4*)(in + idx);
  _Float16* o = out + idx;
  o[0] = (_Float16)v.x; o[1] = (_Float16)v.y; o[2] = (_Float16)v.z; o[3] = (_Float16)v.w;
}

// transpose+convert: in[b][d][x] (f32) -> out[b][x][d] (f16), optional per-d scale
// grid: (X/32, D/32, B), block (32,8)
__global__ void k_transpose_half(const float* __restrict__ in, const float* __restrict__ scale,
                                 _Float16* __restrict__ out, int X) {
  __shared__ float tile[32][33];
  const int b  = blockIdx.z;
  const int x0 = blockIdx.x * 32;
  const int d0 = blockIdx.y * 32;
  const int tx = threadIdx.x, ty = threadIdx.y;
#pragma unroll
  for (int k = 0; k < 4; ++k) {
    int d = d0 + ty + 8 * k;
    float s = scale ? scale[d] : 1.0f;
    tile[ty + 8 * k][tx] = in[((size_t)b * D_ + d) * X + (x0 + tx)] * s;
  }
  __syncthreads();
#pragma unroll
  for (int k = 0; k < 4; ++k) {
    int x = x0 + ty + 8 * k;
    out[((size_t)b * X + x) * D_ + (d0 + tx)] = (_Float16)tile[tx][ty + 8 * k];
  }
}

// bias projections: out[b][x] = sum_d X[b][d][x] * w[d]   (coalesced over x)
__global__ void k_proj(const float* __restrict__ X, const float* __restrict__ w,
                       float* __restrict__ out, int L) {
  const int idx = blockIdx.x * blockDim.x + threadIdx.x;
  const int b = idx / L, p = idx % L;
  const float* Xb = X + (size_t)b * D_ * L + p;
  float s = 0.0f;
#pragma unroll 8
  for (int d = 0; d < D_; ++d) s += Xb[(size_t)d * L] * w[d];
  out[idx] = s;
}

// ---------------- GEMM 1: S[b][i][j] = c1[i] + q2[j] + (Cb*w3) . Qb ----------
__global__ void k_gemm_S(const _Float16* __restrict__ CwT, const _Float16* __restrict__ QT,
                         const float* __restrict__ c1, const float* __restrict__ q2,
                         float* __restrict__ S) {
  const int wave = blockIdx.x * (blockDim.x >> 5) + (threadIdx.x >> 5);
  const int JT = LQ_ / 16, IT = LC_ / 16;
  const int b  = wave / (IT * JT);
  const int r  = wave % (IT * JT);
  const int it = r / JT, jt = r % JT;
  const _Float16* Ab = CwT + ((size_t)b * LC_ + it * 16) * D_;
  const _Float16* Bb = QT  + ((size_t)b * LQ_ + jt * 16) * D_;
  v8f acc = {};
#pragma unroll
  for (int k0 = 0; k0 < D_; k0 += 32)
    acc = wmma16(load_A16(Ab + k0, D_), load_B16(Bb + k0, D_), acc);
  const int lane = threadIdx.x & 31;
  const int j  = jt * 16 + (lane & 15);
  const int ib = it * 16 + ((lane >> 4) << 3);
  const float q2v = q2[b * LQ_ + j];
#pragma unroll
  for (int rr = 0; rr < 8; ++rr)
    S[((size_t)b * LC_ + ib + rr) * LQ_ + j] = acc[rr] + c1[b * LC_ + ib + rr] + q2v;
}

// ---------------- Row softmax (axis=j) -> P1[b][i][j] f16 --------------------
__global__ void k_row_softmax(const float* __restrict__ S, const uint8_t* __restrict__ qmask,
                              _Float16* __restrict__ P1) {
  const int wave = blockIdx.x * (blockDim.x >> 5) + (threadIdx.x >> 5);
  const int lane = threadIdx.x & 31;
  const int b = wave / LC_, i = wave % LC_;
  const float* row = S + ((size_t)b * LC_ + i) * LQ_;
  const uint8_t* qm = qmask + b * LQ_;
  float v[LQ_ / 32];
  float m = NEG_INF;
#pragma unroll
  for (int t = 0; t < LQ_ / 32; ++t) {
    float s = qm[lane + t * 32] ? row[lane + t * 32] : NEG_INF;
    v[t] = s; m = fmaxf(m, s);
  }
#pragma unroll
  for (int off = 16; off >= 1; off >>= 1) m = fmaxf(m, __shfl_xor(m, off, 32));
  float sum = 0.0f;
#pragma unroll
  for (int t = 0; t < LQ_ / 32; ++t) {
    float e = (v[t] <= 0.5f * NEG_INF) ? 0.0f : __expf(v[t] - m);
    v[t] = e; sum += e;
  }
#pragma unroll
  for (int off = 16; off >= 1; off >>= 1) sum += __shfl_xor(sum, off, 32);
  const float inv = sum > 0.0f ? 1.0f / sum : 0.0f;
  _Float16* out = P1 + ((size_t)b * LC_ + i) * LQ_;
#pragma unroll
  for (int t = 0; t < LQ_ / 32; ++t) out[lane + t * 32] = (_Float16)(v[t] * inv);
}

// ---------------- Column softmax stats (axis=i) ------------------------------
__global__ void k_col_stats(const float* __restrict__ S, const uint8_t* __restrict__ cmask,
                            float* __restrict__ cmax, float* __restrict__ cinv) {
  const int idx = blockIdx.x * blockDim.x + threadIdx.x;   // b*LQ + j
  const int b = idx / LQ_, j = idx % LQ_;
  const float* Sb = S + (size_t)b * LC_ * LQ_ + j;
  const uint8_t* cm = cmask + b * LC_;
  float m = NEG_INF;
  for (int i = 0; i < LC_; ++i)
    m = fmaxf(m, cm[i] ? Sb[(size_t)i * LQ_] : NEG_INF);
  float sum = 0.0f;
  for (int i = 0; i < LC_; ++i)
    if (cm[i]) sum += __expf(Sb[(size_t)i * LQ_] - m);
  cmax[idx] = m;
  cinv[idx] = sum > 0.0f ? 1.0f / sum : 0.0f;
}

// Normalize + transpose: P2T[b][j][i] f16.  grid (LQ/32, LC/32, B), block (32,8)
__global__ void k_col_normT(const float* __restrict__ S, const uint8_t* __restrict__ cmask,
                            const float* __restrict__ cmax, const float* __restrict__ cinv,
                            _Float16* __restrict__ P2T) {
  __shared__ float tile[32][33];
  const int b = blockIdx.z;
  const int j0 = blockIdx.x * 32, i0 = blockIdx.y * 32;
  const int tx = threadIdx.x, ty = threadIdx.y;
#pragma unroll
  for (int k = 0; k < 4; ++k) {
    int i = i0 + ty + 8 * k, j = j0 + tx;
    float s = S[((size_t)b * LC_ + i) * LQ_ + j];
    float p = cmask[b * LC_ + i] ? __expf(s - cmax[b * LQ_ + j]) * cinv[b * LQ_ + j] : 0.0f;
    tile[ty + 8 * k][tx] = p;
  }
  __syncthreads();
#pragma unroll
  for (int k = 0; k < 4; ++k) {
    int j = j0 + ty + 8 * k, i = i0 + tx;
    P2T[((size_t)b * LQ_ + j) * LC_ + i] = (_Float16)tile[tx][ty + 8 * k];
  }
}

// ---------------- GEMM 2: Tt[b][d][j] = (S2^T Cb)^T --------------------------
__global__ void k_gemm_T(const _Float16* __restrict__ P2T, const _Float16* __restrict__ Ch,
                         _Float16* __restrict__ Tt) {
  const int wave = blockIdx.x * (blockDim.x >> 5) + (threadIdx.x >> 5);
  const int DT = D_ / 16, JT = LQ_ / 16;
  const int b  = wave / (JT * DT);
  const int r  = wave % (JT * DT);
  const int jt = r / DT, dt = r % DT;
  const _Float16* Ab = P2T + ((size_t)b * LQ_ + jt * 16) * LC_;
  const _Float16* Bb = Ch  + ((size_t)b * D_  + dt * 16) * LC_;
  v8f acc = {};
#pragma unroll 8
  for (int k0 = 0; k0 < LC_; k0 += 32)
    acc = wmma16(load_A16(Ab + k0, LC_), load_B16(Bb + k0, LC_), acc);
  const int lane = threadIdx.x & 31;
  const int d  = dt * 16 + (lane & 15);
  const int jb = jt * 16 + ((lane >> 4) << 3);
  v8h o;
#pragma unroll
  for (int rr = 0; rr < 8; ++rr) o[rr] = (_Float16)acc[rr];
  *(v8h*)(Tt + ((size_t)b * D_ + d) * LQ_ + jb) = o;   // transposed store: contiguous/lane
}

// ---------------- GEMM 3 + epilogue: out = [Cb, A, Cb*A, Cb*Bt] (transposed) -
__global__ void k_gemm_out(const _Float16* __restrict__ P1, const _Float16* __restrict__ Qh,
                           const _Float16* __restrict__ Tt, const float* __restrict__ C,
                           float* __restrict__ out) {
  const int wave = blockIdx.x * (blockDim.x >> 5) + (threadIdx.x >> 5);
  const int DT = D_ / 16, IT = LC_ / 16;
  const int b  = wave / (IT * DT);
  const int r  = wave % (IT * DT);
  const int it = r / DT, dt = r % DT;
  const _Float16* Ab = P1 + ((size_t)b * LC_ + it * 16) * LQ_;
  const _Float16* Bq = Qh + ((size_t)b * D_  + dt * 16) * LQ_;
  const _Float16* Bt = Tt + ((size_t)b * D_  + dt * 16) * LQ_;
  v8f accA = {}, accB = {};
#pragma unroll
  for (int k0 = 0; k0 < LQ_; k0 += 32) {
    v16h a = load_A16(Ab + k0, LQ_);
    accA = wmma16(a, load_B16(Bq + k0, LQ_), accA);
    accB = wmma16(a, load_B16(Bt + k0, LQ_), accB);
  }
  const int lane = threadIdx.x & 31;
  const int d  = dt * 16 + (lane & 15);
  const int ib = it * 16 + ((lane >> 4) << 3);
  v8f cvec = *(const v8f*)(C + ((size_t)b * D_ + d) * LC_ + ib);
  v8f p2, p3;
#pragma unroll
  for (int rr = 0; rr < 8; ++rr) { p2[rr] = cvec[rr] * accA[rr]; p3[rr] = cvec[rr] * accB[rr]; }
  float* ob = out + (size_t)b * 4 * D_ * LC_;
  *(v8f*)(ob + (size_t)(0 * D_ + d) * LC_ + ib) = cvec;   // Cb
  *(v8f*)(ob + (size_t)(1 * D_ + d) * LC_ + ib) = accA;   // A
  *(v8f*)(ob + (size_t)(2 * D_ + d) * LC_ + ib) = p2;     // Cb*A
  *(v8f*)(ob + (size_t)(3 * D_ + d) * LC_ + ib) = p3;     // Cb*Bt
}

// ---------------- Host launcher ---------------------------------------------
extern "C" void kernel_launch(void* const* d_in, const int* in_sizes, int n_in,
                              void* d_out, int out_size, void* d_ws, size_t ws_size,
                              hipStream_t stream) {
  (void)in_sizes; (void)n_in; (void)out_size; (void)ws_size;
  const float*   C     = (const float*)d_in[0];
  const float*   Q     = (const float*)d_in[1];
  const uint8_t* cmask = (const uint8_t*)d_in[2];
  const uint8_t* qmask = (const uint8_t*)d_in[3];
  const float*   w     = (const float*)d_in[4];
  float* out = (float*)d_out;

  char* ws = (char*)d_ws; size_t off = 0;
  auto carve = [&](size_t bytes) -> char* {
    char* p = ws + off; off += (bytes + 255) & ~(size_t)255; return p;
  };
  float*    S    = (float*)   carve((size_t)B_ * LC_ * LQ_ * 4);
  _Float16* P1   = (_Float16*)carve((size_t)B_ * LC_ * LQ_ * 2);
  _Float16* P2T  = (_Float16*)carve((size_t)B_ * LQ_ * LC_ * 2);
  _Float16* Ch   = (_Float16*)carve((size_t)B_ * D_  * LC_ * 2);
  _Float16* CwT  = (_Float16*)carve((size_t)B_ * LC_ * D_  * 2);
  _Float16* Qh   = (_Float16*)carve((size_t)B_ * D_  * LQ_ * 2);
  _Float16* QT   = (_Float16*)carve((size_t)B_ * LQ_ * D_  * 2);
  _Float16* Tt   = (_Float16*)carve((size_t)B_ * D_  * LQ_ * 2);
  float*    c1   = (float*)   carve((size_t)B_ * LC_ * 4);
  float*    q2   = (float*)   carve((size_t)B_ * LQ_ * 4);
  float*    cmax = (float*)   carve((size_t)B_ * LQ_ * 4);
  float*    cinv = (float*)   carve((size_t)B_ * LQ_ * 4);

  // --- prep: conversions, transposes, bias projections
  k_conv_half<<<(B_*D_*LC_)/(256*4), 256, 0, stream>>>(C, Ch);
  k_conv_half<<<(B_*D_*LQ_)/(256*4), 256, 0, stream>>>(Q, Qh);
  k_transpose_half<<<dim3(LC_/32, D_/32, B_), dim3(32, 8), 0, stream>>>(C, w + 2*D_, CwT, LC_);
  k_transpose_half<<<dim3(LQ_/32, D_/32, B_), dim3(32, 8), 0, stream>>>(Q, nullptr,  QT,  LQ_);
  k_proj<<<(B_*LC_)/256, 256, 0, stream>>>(C, w,       c1, LC_);
  k_proj<<<(B_*LQ_)/256, 256, 0, stream>>>(Q, w + D_,  q2, LQ_);

  // --- S = bias + (Cb*w3) Qb^T   (WMMA)
  k_gemm_S<<<(B_*(LC_/16)*(LQ_/16))/8, 256, 0, stream>>>(CwT, QT, c1, q2, S);

  // --- softmaxes
  k_row_softmax<<<(B_*LC_)/8, 256, 0, stream>>>(S, qmask, P1);
  k_col_stats<<<(B_*LQ_)/256, 256, 0, stream>>>(S, cmask, cmax, cinv);
  k_col_normT<<<dim3(LQ_/32, LC_/32, B_), dim3(32, 8), 0, stream>>>(S, cmask, cmax, cinv, P2T);

  // --- T = S2^T Cb (stored transposed)   (WMMA)
  k_gemm_T<<<(B_*(LQ_/16)*(D_/16))/8, 256, 0, stream>>>(P2T, Ch, Tt);

  // --- A = S1 Qb, Bt = S1 T, fused epilogue + transposed output   (WMMA)
  k_gemm_out<<<(B_*(LC_/16)*(D_/16))/8, 256, 0, stream>>>(P1, Qh, Tt, C, out);
}
